// SimpleGCNLayer_39367670235762
// MI455X (gfx1250) — compile-verified, matrix operand order checked by reading, not run on hardware
//
#include <hip/hip_runtime.h>
#include <hip/hip_bf16.h>
#include <stdint.h>

#define B_ 4
#define V_ 10000
#define F_ 128
#define E_ 160000
#define EPS_ 1e-5f

typedef __attribute__((ext_vector_type(16))) __bf16 v16bf;
typedef __attribute__((ext_vector_type(8)))  float  v8f;

__device__ __forceinline__ uint16_t f32_to_bf16_rne(float f) {
    uint32_t u = __float_as_uint(f);
    uint32_t lsb = (u >> 16) & 1u;
    u += 0x7FFFu + lsb;
    return (uint16_t)(u >> 16);
}

// ---- Kernel 0a: X (fp32, B*V*F) -> bf16 workspace --------------------------
__global__ void cvt_x(const float* __restrict__ X, uint16_t* __restrict__ Xbf) {
    int idx = blockIdx.x * blockDim.x + threadIdx.x;
    if (idx < B_ * V_ * F_) Xbf[idx] = f32_to_bf16_rne(X[idx]);
}

// ---- Kernel 0b: pack [W_self ; W_node] (each 128x128, row=out, col=in) into
//      a combined 256x128 bf16 weight block. Rows 0..127 -> H, 128..255 -> M.
__global__ void cvt_w(const float* __restrict__ W_self, const float* __restrict__ W_node,
                      uint16_t* __restrict__ Wbf) {
    int idx = blockIdx.x * blockDim.x + threadIdx.x;
    if (idx >= 256 * F_) return;
    int o = idx / F_;
    int k = idx % F_;
    float w = (o < F_) ? W_self[o * F_ + k] : W_node[(o - F_) * F_ + k];
    Wbf[idx] = f32_to_bf16_rne(w);
}

// ---- Kernel 1: WMMA GEMM. One block (4 waves) = one 16-node tile.
// Wave w computes N columns [w*64, w*64+64): N<128 -> H(+bias), N>=128 -> M.
__global__ __launch_bounds__(128)
void gcn_gemm(const uint16_t* __restrict__ Xbf, const uint16_t* __restrict__ Wbf,
              const float* __restrict__ b_self, float* __restrict__ H,
              float* __restrict__ M) {
    const int lane = threadIdx.x & 31;
    const int wave = threadIdx.x >> 5;
    const int ntile_count = V_ / 16;          // 625
    const int tile = blockIdx.x % ntile_count;
    const int b    = blockIdx.x / ntile_count;
    const int v0   = tile * 16;
    const int r    = lane & 15;               // row within A-tile / col of B,C,D
    const int hi   = lane >> 4;               // lane-half select

    const uint32_t* Xrow = (const uint32_t*)(Xbf + (size_t)(b * V_ + v0 + r) * F_);
    const int nbase = wave * 64;

    v8f acc[4];
#pragma unroll
    for (int i = 0; i < 4; ++i) acc[i] = (v8f){0, 0, 0, 0, 0, 0, 0, 0};

#pragma unroll
    for (int ks = 0; ks < 4; ++ks) {          // K = 128 in 4 steps of 32
        const int kbase = ks * 32;
        union { v16bf v; uint32_t u[8]; } a;
#pragma unroll
        for (int j = 0; j < 8; ++j) {
            const int kp = ((j < 4) ? 0 : 16) + hi * 8 + (j & 3) * 2;
            a.u[j] = Xrow[(kbase + kp) >> 1];
        }
#pragma unroll
        for (int nt = 0; nt < 4; ++nt) {
            const int n = nbase + nt * 16 + r;
            const uint32_t* Wrow = (const uint32_t*)(Wbf + (size_t)n * F_);
            union { v16bf v; uint32_t u[8]; } bb;
#pragma unroll
            for (int j = 0; j < 8; ++j) {
                const int kp = ((j < 4) ? 0 : 16) + hi * 8 + (j & 3) * 2;
                bb.u[j] = Wrow[(kbase + kp) >> 1];
            }
            acc[nt] = __builtin_amdgcn_wmma_f32_16x16x32_bf16(
                false, a.v, false, bb.v, (short)0, acc[nt], false, false);
        }
    }

    // C/D layout: VGPR i -> row m = i + 8*hi; column n = r (+ subtile base)
#pragma unroll
    for (int nt = 0; nt < 4; ++nt) {
        const int n = nbase + nt * 16 + r;
#pragma unroll
        for (int i = 0; i < 8; ++i) {
            const int m = i + hi * 8;
            const size_t row = (size_t)(b * V_ + v0 + m);
            const float val = acc[nt][i];
            if (n < F_) {
                H[row * F_ + n] = val + b_self[n];
            } else {
                M[row * F_ + (n - F_)] = val;
            }
        }
    }
}

// ---- Kernel 2: per-edge scatter-add. Block = 1 edge, 128 channel lanes. ----
__global__ __launch_bounds__(128)
void gcn_scatter(const long long* __restrict__ edges, const float* __restrict__ M,
                 float* __restrict__ H) {
    const int e = blockIdx.x;
    const int c = threadIdx.x;
    const int src = (int)edges[e];
    const int dst = (int)edges[E_ + e];
#pragma unroll
    for (int b = 0; b < B_; ++b) {
        const float val = M[(size_t)(b * V_ + src) * F_ + c];
        atomicAdd(&H[(size_t)(b * V_ + dst) * F_ + c], val);
    }
}

// ---- Kernel 3: BN statistics -----------------------------------------------
__global__ void bn_zero(float* __restrict__ stats) {
    if (threadIdx.x < 2 * F_) stats[threadIdx.x] = 0.0f;
}

__global__ __launch_bounds__(128)
void bn_stats(const float* __restrict__ H, float* __restrict__ stats) {
    const int c = threadIdx.x;                 // channel
    const int rows = B_ * V_;
    const int rpb = (rows + gridDim.x - 1) / gridDim.x;
    const int r0 = blockIdx.x * rpb;
    const int r1 = (r0 + rpb < rows) ? (r0 + rpb) : rows;
    float s = 0.0f, s2 = 0.0f;
    for (int r = r0; r < r1; ++r) {
        const float h = H[(size_t)r * F_ + c];
        s += h;
        s2 += h * h;
    }
    atomicAdd(&stats[c], s);
    atomicAdd(&stats[F_ + c], s2);
}

// ---- Kernel 4: normalize * gamma + beta, ReLU, in place in d_out -----------
__global__ void bn_apply(float* __restrict__ H, const float* __restrict__ stats,
                         const float* __restrict__ gamma, const float* __restrict__ beta) {
    const int idx = blockIdx.x * blockDim.x + threadIdx.x;
    if (idx >= B_ * V_ * F_) return;
    const int c = idx & (F_ - 1);
    const float invN = 1.0f / (float)(B_ * V_);
    const float mean = stats[c] * invN;
    const float var  = stats[F_ + c] * invN - mean * mean;
    float h = (H[idx] - mean) * rsqrtf(var + EPS_);
    h = h * gamma[c] + beta[c];
    H[idx] = fmaxf(h, 0.0f);
}

extern "C" void kernel_launch(void* const* d_in, const int* in_sizes, int n_in,
                              void* d_out, int out_size, void* d_ws, size_t ws_size,
                              hipStream_t stream) {
    (void)in_sizes; (void)n_in; (void)out_size; (void)ws_size;
    const float*     X      = (const float*)d_in[0];
    const long long* edges  = (const long long*)d_in[1];   // int64 (2, E)
    const float*     W_node = (const float*)d_in[2];
    const float*     W_self = (const float*)d_in[3];
    const float*     b_self = (const float*)d_in[4];
    const float*     gamma  = (const float*)d_in[5];
    const float*     beta   = (const float*)d_in[6];
    float*           H      = (float*)d_out;              // (B, V, F_OUT)

    // Workspace layout
    char* ws = (char*)d_ws;
    uint16_t* Xbf  = (uint16_t*)ws;                                    // B*V*F bf16
    uint16_t* Wbf  = (uint16_t*)(ws + (size_t)B_ * V_ * F_ * 2);       // 256*128 bf16
    float*    Mbuf = (float*)(ws + (size_t)B_ * V_ * F_ * 2 + 256 * F_ * 2);
    float*    stats = (float*)((char*)Mbuf + (size_t)B_ * V_ * F_ * 4); // 2*F floats

    const int total = B_ * V_ * F_;
    cvt_x<<<(total + 255) / 256, 256, 0, stream>>>(X, Xbf);
    cvt_w<<<(256 * F_ + 255) / 256, 256, 0, stream>>>(W_self, W_node, Wbf);
    gcn_gemm<<<B_ * (V_ / 16), 128, 0, stream>>>(Xbf, Wbf, b_self, H, Mbuf);
    gcn_scatter<<<E_, 128, 0, stream>>>(edges, Mbuf, H);
    bn_zero<<<1, 256, 0, stream>>>(stats);
    bn_stats<<<400, 128, 0, stream>>>(H, stats);
    bn_apply<<<(total + 255) / 256, 256, 0, stream>>>(H, stats, gamma, beta);
}
